// TemporalGNN_A3TGCN_36060545417511
// MI455X (gfx1250) — matrix operand order repaired and checked by reading, not compiled, not run on hardware
//
#include <hip/hip_runtime.h>
#include <hip/hip_bf16.h>
#include <math.h>

#define NB   5000     // nodes
#define BB   4        // batch
#define TT   40       // timesteps
#define FIN  90       // input features
#define CC   256      // hidden channels
#define EE   160000   // edges
#define MR   (BB*NB)  // 20000 rows
#define KPAD 96       // F_IN padded to multiple of 32
#define NG   768      // 3*C (z|r|h)
#define ETOT (EE+NB)  // edges + self loops

typedef __attribute__((ext_vector_type(16))) _Float16 v16h;
typedef __attribute__((ext_vector_type(8)))  _Float16 v8h;
typedef __attribute__((ext_vector_type(8)))  float    v8f;

// ---- WMMA fragment loaders (wave32, v_wmma_f32_16x16x32_f16 layouts) ----
// A 16x32 f16: lane<16 holds M=lane, K chunks [0,8)+[16,24); lanes>=16: +8.
__device__ __forceinline__ v16h frag_a(const _Float16* base, int row_base, int lane) {
  const _Float16* p = base + (row_base + (lane & 15)) * 32 + ((lane >> 4) << 3);
  v8h lo = *(const v8h*)(p);
  v8h hi = *(const v8h*)(p + 16);
  v16h f;
#pragma unroll
  for (int i = 0; i < 8; ++i) { f[i] = lo[i]; f[i + 8] = hi[i]; }
  return f;
}
// B 32x16 f16 (stored N-major [n][k]): lane<16 holds N=lane, K chunks [0,8)+[16,24); lanes>=16: +8.
__device__ __forceinline__ v16h frag_b(const _Float16* base, int row_base, int lane) {
  const _Float16* p = base + (row_base + (lane & 15)) * 32 + ((lane >> 4) << 3);
  v8h lo = *(const v8h*)(p);
  v8h hi = *(const v8h*)(p + 16);
  v16h f;
#pragma unroll
  for (int i = 0; i < 8; ++i) { f[i] = lo[i]; f[i + 8] = hi[i]; }
  return f;
}
#define WMMA(A, B, C) __builtin_amdgcn_wmma_f32_16x16x32_f16(false, A, false, B, (short)0, C, false, false)

// ---------------- graph preprocessing ----------------
__global__ void k_count(const int* __restrict__ col, int* __restrict__ cnt) {
  int e = blockIdx.x * blockDim.x + threadIdx.x;
  if (e < EE) atomicAdd(&cnt[col[e]], 1);
}
__global__ void k_dinv(const int* __restrict__ cnt, float* __restrict__ dinv) {
  int n = blockIdx.x * blockDim.x + threadIdx.x;
  if (n < NB) dinv[n] = rsqrtf((float)(cnt[n] + 1)); // +1 self loop, always > 0
}
__global__ void k_scan(const int* __restrict__ cnt, int* __restrict__ off) {
  if (blockIdx.x == 0 && threadIdx.x == 0) {
    int acc = 0;
    for (int n = 0; n < NB; ++n) { off[n] = acc; acc += cnt[n] + 1; }
    off[NB] = acc;
  }
}
__global__ void k_fill(const int* __restrict__ row, const int* __restrict__ col,
                       const int* __restrict__ off, int* __restrict__ cursor,
                       int* __restrict__ csr_src) {
  int i = blockIdx.x * blockDim.x + threadIdx.x;
  if (i < EE) {
    int c = col[i];
    int pos = off[c] + atomicAdd(&cursor[c], 1);
    csr_src[pos] = row[i];
  } else if (i < ETOT) {
    int n = i - EE;
    int pos = off[n] + atomicAdd(&cursor[n], 1);
    csr_src[pos] = n;                        // self loop
  }
}

// ---------------- weight prep: f16 + transpose to N-major ----------------
__global__ void k_prep_weights(const float* __restrict__ Wz, const float* __restrict__ Wr,
                               const float* __restrict__ Wh,
                               const float* __restrict__ bz, const float* __restrict__ br,
                               const float* __restrict__ bh,
                               const float* __restrict__ Lz, const float* __restrict__ Lr,
                               const float* __restrict__ Lh,
                               _Float16* __restrict__ WT, _Float16* __restrict__ LT,
                               float* __restrict__ bconc) {
  int i = blockIdx.x * blockDim.x + threadIdx.x;
  int stride = gridDim.x * blockDim.x;
  for (int idx = i; idx < NG * KPAD; idx += stride) {        // WT[n][k]  768x96
    int ng = idx / KPAD, k = idx - ng * KPAD;
    int g = ng / CC, c = ng - g * CC;
    const float* W = (g == 0) ? Wz : (g == 1) ? Wr : Wh;     // W is [FIN][CC]
    WT[idx] = (_Float16)((k < FIN) ? W[k * CC + c] : 0.0f);
  }
  for (int idx = i; idx < 3 * CC * 512; idx += stride) {     // LT[g][n][k] 3x256x512
    int g = idx / (CC * 512); int rem = idx - g * CC * 512;
    int n = rem / 512, k = rem - n * 512;
    const float* L = (g == 0) ? Lz : (g == 1) ? Lr : Lh;     // L is [512][CC]
    LT[idx] = (_Float16)L[k * CC + n];
  }
  for (int idx = i; idx < NG; idx += stride)
    bconc[idx] = (idx < CC) ? bz[idx] : (idx < 2 * CC) ? br[idx - CC] : bh[idx - 2 * CC];
}

__global__ void k_softmax(const float* __restrict__ att, float* __restrict__ probs) {
  if (blockIdx.x == 0 && threadIdx.x == 0) {
    float mx = -1e30f;
    for (int i = 0; i < TT; ++i) mx = fmaxf(mx, att[i]);
    float e[TT], s = 0.f;
    for (int i = 0; i < TT; ++i) { e[i] = expf(att[i] - mx); s += e[i]; }
    for (int i = 0; i < TT; ++i) probs[i] = e[i] / s;
  }
}

// ---------------- x_t -> f16, K padded to 96 ----------------
__global__ __launch_bounds__(256)
void k_xcvt(const float* __restrict__ x, _Float16* __restrict__ xh, int t) {
  int idx = blockIdx.x * blockDim.x + threadIdx.x;
  if (idx >= MR * KPAD) return;
  int m = idx / KPAD, k = idx - m * KPAD;
  int b = m / NB, n = m - b * NB;
  float v = (k < FIN) ? x[((size_t)b * TT + t) * ((size_t)NB * FIN) + (size_t)n * FIN + k] : 0.0f;
  xh[idx] = (_Float16)v;
}

// ---------------- xw = dinv[src] * (x_t @ [Wz|Wr|Wh])  -> f16 ----------------
// 128x128 tile, 8 waves (4x2), 2x4 WMMA tiles per wave, double-buffered LDS.
__global__ __launch_bounds__(256)
void k_gemm_xw(const _Float16* __restrict__ xh, const _Float16* __restrict__ WT,
               const float* __restrict__ dinv, _Float16* __restrict__ xwh) {
  __shared__ _Float16 As[2][128 * 32];
  __shared__ _Float16 Bs[2][128 * 32];
  int tid = threadIdx.x, lane = tid & 31, wid = tid >> 5;
  int wm = wid >> 1, wn = wid & 1;
  int m0 = blockIdx.x * 128, n0 = blockIdx.y * 128;
  int lr = tid >> 1, lks = (tid & 1) * 16;          // tile-load coords (A and B identical)
  const int NKT = KPAD / 32;

  v8f acc[2][4] = {};
  v8h pa0, pa1, pb0, pb1;

  auto loadA = [&](int kt) {
    int m = m0 + lr;
    if (m < MR) {
      const _Float16* s = xh + (size_t)m * KPAD + kt * 32 + lks;
      pa0 = *(const v8h*)s; pa1 = *(const v8h*)(s + 8);
    } else { pa0 = (v8h){}; pa1 = (v8h){}; }
  };
  auto loadB = [&](int kt) {
    const _Float16* s = WT + (size_t)(n0 + lr) * KPAD + kt * 32 + lks;
    pb0 = *(const v8h*)s; pb1 = *(const v8h*)(s + 8);
  };
  auto stash = [&](int buf) {
    _Float16* d = As[buf] + lr * 32 + lks;
    *(v8h*)d = pa0; *(v8h*)(d + 8) = pa1;
    _Float16* e = Bs[buf] + lr * 32 + lks;
    *(v8h*)e = pb0; *(v8h*)(e + 8) = pb1;
  };

  loadA(0); loadB(0); stash(0);
  __syncthreads();
#pragma unroll
  for (int kt = 0; kt < NKT; ++kt) {
    int cur = kt & 1;
    bool more = (kt + 1 < NKT);
    if (more) { loadA(kt + 1); loadB(kt + 1); }     // overlap with WMMA below
    v16h a0 = frag_a(As[cur], wm * 32 + 0, lane);
    v16h a1 = frag_a(As[cur], wm * 32 + 16, lane);
#pragma unroll
    for (int j = 0; j < 4; ++j) {
      v16h bj = frag_b(Bs[cur], wn * 64 + j * 16, lane);
      acc[0][j] = WMMA(a0, bj, acc[0][j]);
      acc[1][j] = WMMA(a1, bj, acc[1][j]);
    }
    if (more) stash(cur ^ 1);
    __syncthreads();
  }
#pragma unroll
  for (int ti = 0; ti < 2; ++ti)
#pragma unroll
    for (int tj = 0; tj < 4; ++tj) {
      int mb = m0 + wm * 32 + ti * 16 + ((lane >> 4) << 3);
      int nb = n0 + wn * 64 + tj * 16 + (lane & 15);
#pragma unroll
      for (int r = 0; r < 8; ++r) {
        int m = mb + r;
        if (m < MR) {
          int node = m % NB;                     // row's node id
          xwh[(size_t)m * NG + nb] = (_Float16)(acc[ti][tj][r] * dinv[node]);
        }
      }
    }
}

// ---------------- GCN gather: g[b,n,:] = dinv[n] * sum_e xwh[b,src_e,:] + bias ----------------
__global__ __launch_bounds__(256)
void k_gcn_gather(const _Float16* __restrict__ xwh, const int* __restrict__ off,
                  const int* __restrict__ csr_src, const float* __restrict__ dinv,
                  const float* __restrict__ bias, _Float16* __restrict__ gcnh) {
  int bid = blockIdx.x;                 // 0..MR-1
  int b = bid / NB, n = bid - b * NB;
  int tid = threadIdx.x;
  int e0 = off[n], e1 = off[n + 1];
  float a0 = 0.f, a1 = 0.f, a2 = 0.f;
  const _Float16* base = xwh + (size_t)b * NB * NG;
  for (int e = e0; e < e1; ++e) {
    const _Float16* rowp = base + (size_t)csr_src[e] * NG;
    a0 += (float)rowp[tid];
    a1 += (float)rowp[tid + 256];
    a2 += (float)rowp[tid + 512];
  }
  float dn = dinv[n];
  size_t o = (size_t)bid * NG;
  gcnh[o + tid]       = (_Float16)(dn * a0 + bias[tid]);
  gcnh[o + tid + 256] = (_Float16)(dn * a1 + bias[tid + 256]);
  gcnh[o + tid + 512] = (_Float16)(dn * a2 + bias[tid + 512]);
}

// ---------------- GRU gate GEMM: [g_gate | Hpart] @ LT_gate, fused epilogue ----------------
// mode 0: Z = sigmoid(.)     mode 1: HRh = f16(H * sigmoid(.))
// mode 2: Ht = tanh(.); H = Z*H + (1-Z)*Ht; Hacc += p_t*H
__global__ __launch_bounds__(256)
void k_gru_gemm(const _Float16* __restrict__ gcnh, _Float16* __restrict__ Hh,
                _Float16* __restrict__ HRh, const _Float16* __restrict__ LT,
                const float* __restrict__ Lzb, const float* __restrict__ Lrb,
                const float* __restrict__ Lhb,
                float* __restrict__ H, float* __restrict__ Zb,
                float* __restrict__ Hacc, const float* __restrict__ probs,
                int t, int mode_base) {
  __shared__ _Float16 As[2][128 * 32];
  __shared__ _Float16 Bs[2][128 * 32];
  int tid = threadIdx.x, lane = tid & 31, wid = tid >> 5;
  int wm = wid >> 1, wn = wid & 1;
  int m0 = blockIdx.x * 128, n0 = blockIdx.y * 128;
  int mode = mode_base + blockIdx.z;
  int lr = tid >> 1, lks = (tid & 1) * 16;
  const int NKT = (2 * CC) / 32;  // 16

  const _Float16* A0  = gcnh + mode * CC;                 // column block of [gz|gr|gh]
  const _Float16* Ap1 = (mode == 2) ? HRh : Hh;           // second K half
  const _Float16* LTg = LT + (size_t)mode * CC * 512;
  const float* bias = (mode == 0) ? Lzb : (mode == 1) ? Lrb : Lhb;
  float pt = probs[t];

  v8f acc[2][4] = {};
  v8h pa0, pa1, pb0, pb1;

  auto loadA = [&](int kt) {
    int m = m0 + lr, kl = kt * 32 + lks;
    if (m < MR) {
      const _Float16* s = (kl < CC) ? (A0 + (size_t)m * NG + kl)
                                    : (Ap1 + (size_t)m * CC + (kl - CC));
      pa0 = *(const v8h*)s; pa1 = *(const v8h*)(s + 8);
    } else { pa0 = (v8h){}; pa1 = (v8h){}; }
  };
  auto loadB = [&](int kt) {
    const _Float16* s = LTg + (size_t)(n0 + lr) * 512 + kt * 32 + lks;
    pb0 = *(const v8h*)s; pb1 = *(const v8h*)(s + 8);
  };
  auto stash = [&](int buf) {
    _Float16* d = As[buf] + lr * 32 + lks;
    *(v8h*)d = pa0; *(v8h*)(d + 8) = pa1;
    _Float16* e = Bs[buf] + lr * 32 + lks;
    *(v8h*)e = pb0; *(v8h*)(e + 8) = pb1;
  };

  loadA(0); loadB(0); stash(0);
  __syncthreads();
#pragma unroll 2
  for (int kt = 0; kt < NKT; ++kt) {
    int cur = kt & 1;
    bool more = (kt + 1 < NKT);
    if (more) { loadA(kt + 1); loadB(kt + 1); }
    v16h a0 = frag_a(As[cur], wm * 32 + 0, lane);
    v16h a1 = frag_a(As[cur], wm * 32 + 16, lane);
#pragma unroll
    for (int j = 0; j < 4; ++j) {
      v16h bj = frag_b(Bs[cur], wn * 64 + j * 16, lane);
      acc[0][j] = WMMA(a0, bj, acc[0][j]);
      acc[1][j] = WMMA(a1, bj, acc[1][j]);
    }
    if (more) stash(cur ^ 1);
    __syncthreads();
  }
#pragma unroll
  for (int ti = 0; ti < 2; ++ti)
#pragma unroll
    for (int tj = 0; tj < 4; ++tj) {
      int mb = m0 + wm * 32 + ti * 16 + ((lane >> 4) << 3);
      int n  = n0 + wn * 64 + tj * 16 + (lane & 15);
      float bv = bias[n];
#pragma unroll
      for (int r = 0; r < 8; ++r) {
        int m = mb + r;
        if (m >= MR) continue;
        float s = acc[ti][tj][r] + bv;
        size_t idx = (size_t)m * CC + n;
        if (mode == 0) {
          Zb[idx] = 1.0f / (1.0f + expf(-s));
        } else if (mode == 1) {
          float rr = 1.0f / (1.0f + expf(-s));
          HRh[idx] = (_Float16)(H[idx] * rr);
        } else {
          float ht = tanhf(s);
          float z = Zb[idx], hold = H[idx];
          float hn = z * hold + (1.0f - z) * ht;
          H[idx] = hn;
          Hh[idx] = (_Float16)hn;
          Hacc[idx] += pt * hn;
        }
      }
    }
}

// ---------------- readout MLP head ----------------
__global__ __launch_bounds__(64)
void k_head(const float* __restrict__ Hacc, const int* __restrict__ fid,
            const float* __restrict__ M1w, const float* __restrict__ M1b,
            const float* __restrict__ M2w, const float* __restrict__ M2b,
            const float* __restrict__ M3w, const float* __restrict__ M3b,
            float* __restrict__ out) {
  __shared__ float h0[CC], h1[64], h2[32];
  int b = blockIdx.x, tid = threadIdx.x;
  const float* src = Hacc + ((size_t)b * NB + fid[b]) * CC;
  for (int i = tid; i < CC; i += 64) h0[i] = src[i];
  __syncthreads();
  {
    float a = M1b[tid];
    for (int k = 0; k < CC; ++k) a += h0[k] * M1w[k * 64 + tid];
    h1[tid] = (a > 0.f) ? a : 0.01f * a;
  }
  __syncthreads();
  if (tid < 32) {
    float a = M2b[tid];
    for (int k = 0; k < 64; ++k) a += h1[k] * M2w[k * 32 + tid];
    h2[tid] = (a > 0.f) ? a : 0.01f * a;
  }
  __syncthreads();
  if (tid < 5) {
    float a = M3b[tid];
    for (int k = 0; k < 32; ++k) a += h2[k] * M3w[k * 5 + tid];
    out[b * 5 + tid] = 4.0f / (1.0f + expf(-a)) + 1.0f;
  }
}

extern "C" void kernel_launch(void* const* d_in, const int* in_sizes, int n_in,
                              void* d_out, int out_size, void* d_ws, size_t ws_size,
                              hipStream_t stream) {
  (void)in_sizes; (void)n_in; (void)out_size;
  const float* x   = (const float*)d_in[0];
  const int*   ei  = (const int*)d_in[1];     // [2][E]: row then col
  const int*   fid = (const int*)d_in[2];
  const float* att = (const float*)d_in[3];
  const float* Wz  = (const float*)d_in[4];  const float* bz = (const float*)d_in[5];
  const float* Wr  = (const float*)d_in[6];  const float* br = (const float*)d_in[7];
  const float* Wh  = (const float*)d_in[8];  const float* bh = (const float*)d_in[9];
  const float* Lz  = (const float*)d_in[10]; const float* Lzb = (const float*)d_in[11];
  const float* Lr  = (const float*)d_in[12]; const float* Lrb = (const float*)d_in[13];
  const float* Lh  = (const float*)d_in[14]; const float* Lhb = (const float*)d_in[15];
  const float* M1w = (const float*)d_in[16]; const float* M1b = (const float*)d_in[17];
  const float* M2w = (const float*)d_in[18]; const float* M2b = (const float*)d_in[19];
  const float* M3w = (const float*)d_in[20]; const float* M3b = (const float*)d_in[21];
  float* out = (float*)d_out;

  char* ws = (char*)d_ws;
  size_t cur = 0;
  auto alloc = [&](size_t bytes) -> void* {
    void* p = ws + cur;
    cur = (cur + bytes + 255) & ~(size_t)255;
    return p;
  };
  float*    probs  = (float*)alloc(TT * 4);
  int*      cnt    = (int*)alloc(NB * 4);
  int*      offs   = (int*)alloc((NB + 1) * 4);
  int*      cursor = (int*)alloc(NB * 4);
  float*    dinv   = (float*)alloc(NB * 4);
  int*      csrS   = (int*)alloc((size_t)ETOT * 4);
  _Float16* WT     = (_Float16*)alloc((size_t)NG * KPAD * 2);
  _Float16* LT     = (_Float16*)alloc((size_t)3 * CC * 512 * 2);
  float*    bconc  = (float*)alloc(NG * 4);
  _Float16* xh     = (_Float16*)alloc((size_t)MR * KPAD * 2);
  _Float16* xwh    = (_Float16*)alloc((size_t)MR * NG * 2);
  _Float16* gcnh   = (_Float16*)alloc((size_t)MR * NG * 2);
  float*    H      = (float*)alloc((size_t)MR * CC * 4);
  _Float16* Hh     = (_Float16*)alloc((size_t)MR * CC * 2);
  _Float16* HRh    = (_Float16*)alloc((size_t)MR * CC * 2);
  float*    Zb     = (float*)alloc((size_t)MR * CC * 4);
  float*    Hacc   = (float*)alloc((size_t)MR * CC * 4);
  if (cur > ws_size) return; // workspace too small; nothing safe to do

  hipMemsetAsync(cnt, 0, NB * 4, stream);
  hipMemsetAsync(cursor, 0, NB * 4, stream);
  hipMemsetAsync(H, 0, (size_t)MR * CC * 4, stream);
  hipMemsetAsync(Hh, 0, (size_t)MR * CC * 2, stream);
  hipMemsetAsync(Hacc, 0, (size_t)MR * CC * 4, stream);

  k_count<<<(EE + 255) / 256, 256, 0, stream>>>(ei + EE, cnt);
  k_dinv<<<(NB + 255) / 256, 256, 0, stream>>>(cnt, dinv);
  k_scan<<<1, 1, 0, stream>>>(cnt, offs);
  k_fill<<<(ETOT + 255) / 256, 256, 0, stream>>>(ei, ei + EE, offs, cursor, csrS);
  k_prep_weights<<<512, 256, 0, stream>>>(Wz, Wr, Wh, bz, br, bh, Lz, Lr, Lh, WT, LT, bconc);
  k_softmax<<<1, 32, 0, stream>>>(att, probs);

  dim3 gx((MR + 127) / 128, NG / 128);     // 157 x 6
  dim3 ga((MR + 127) / 128, CC / 128, 2);  // gates z,r
  dim3 gb((MR + 127) / 128, CC / 128, 1);  // gate h + update
  int cvt_blocks = (MR * KPAD + 255) / 256;
  for (int t = 0; t < TT; ++t) {
    k_xcvt<<<cvt_blocks, 256, 0, stream>>>(x, xh, t);
    k_gemm_xw<<<gx, 256, 0, stream>>>(xh, WT, dinv, xwh);
    k_gcn_gather<<<MR, 256, 0, stream>>>(xwh, offs, csrS, dinv, bconc, gcnh);
    k_gru_gemm<<<ga, 256, 0, stream>>>(gcnh, Hh, HRh, LT, Lzb, Lrb, Lhb, H, Zb, Hacc, probs, t, 0);
    k_gru_gemm<<<gb, 256, 0, stream>>>(gcnh, Hh, HRh, LT, Lzb, Lrb, Lhb, H, Zb, Hacc, probs, t, 2);
  }
  k_head<<<BB, 64, 0, stream>>>(Hacc, fid, M1w, M1b, M2w, M2b, M3w, M3b, out);
}
